// DESOM_2808908612213
// MI455X (gfx1250) — matrix operand order repaired and checked by reading, not compiled, not run on hardware
//
#include <hip/hip_runtime.h>
#include <hip/hip_bf16.h>
#include <stdint.h>

// ---------------------------------------------------------------------------
// DESOM forward for MI455X (gfx1250, wave32, WMMA).
// bf16 GEMMs via v_wmma_f32_16x16x32_bf16, double-buffered LDS pipeline,
// vectorized (b128 load / ds b64 store) staging on full tiles.
// ---------------------------------------------------------------------------

typedef __attribute__((ext_vector_type(16))) __bf16 v16bf;
typedef __attribute__((ext_vector_type(8)))  float  v8f;

union FragAB { v16bf v; uint32_t u[8]; };
union FragC  { v8f   v; float    f[8]; };

__device__ __forceinline__ uint32_t bf16_1(float f) {
  uint32_t u = __float_as_uint(f);
  u += 0x7FFFu + ((u >> 16) & 1u);   // RNE
  return u >> 16;
}
__device__ __forceinline__ uint32_t bf16_2(float lo, float hi) {
  return bf16_1(lo) | (bf16_1(hi) << 16);
}

#define BM 128
#define BN 128
#define BK 32
#define LDSS 36   // padded stride (bf16 units); 36r+4c keeps 8B alignment

// Stage one (k-chunk) pair of tiles into the given LDS buffers.
// MODE 0/1: Wm is [K,N];  MODE 2: Wm is [N,K] (prototypes).
template <int MODE>
__device__ __forceinline__ void stage_tiles(
    const float* __restrict__ A, const float* __restrict__ Wm,
    uint16_t* __restrict__ Asb, uint16_t* __restrict__ Bsb,
    int tid, int m0, int n0, int k0, int N, int K) {
  const bool fullK = (k0 + BK) <= K;
  const bool fullN = (n0 + BN) <= N;

  // ---------------- A tile: [BM][BK] ----------------
  if (fullK) {
#pragma unroll
    for (int i = 0; i < 4; ++i) {
      int flat4 = i * 256 + tid;          // 0..1023, 4 elems each
      int r = flat4 >> 3, c4 = (flat4 & 7) * 4;
      const float4 f = *reinterpret_cast<const float4*>(
          A + (size_t)(m0 + r) * K + (k0 + c4));
      uint2 p = make_uint2(bf16_2(f.x, f.y), bf16_2(f.z, f.w));
      *reinterpret_cast<uint2*>(&Asb[r * LDSS + c4]) = p;
    }
  } else {
#pragma unroll
    for (int i = 0; i < 16; ++i) {
      int flat = i * 256 + tid;
      int r = flat >> 5, c = flat & 31;
      int gk = k0 + c;
      float v = (gk < K) ? A[(size_t)(m0 + r) * K + gk] : 0.0f;
      Asb[r * LDSS + c] = (uint16_t)bf16_1(v);
    }
  }

  // ---------------- B tile, transposed in LDS: Bs[n][k] ----------------
  if (MODE == 2) {
    if (fullK && fullN) {
#pragma unroll
      for (int i = 0; i < 4; ++i) {
        int flat4 = i * 256 + tid;
        int nn = flat4 >> 3, k4 = (flat4 & 7) * 4;
        const float4 f = *reinterpret_cast<const float4*>(
            Wm + (size_t)(n0 + nn) * K + (k0 + k4));
        uint2 p = make_uint2(bf16_2(f.x, f.y), bf16_2(f.z, f.w));
        *reinterpret_cast<uint2*>(&Bsb[nn * LDSS + k4]) = p;
      }
    } else {
#pragma unroll
      for (int i = 0; i < 16; ++i) {
        int flat = i * 256 + tid;
        int nn = flat >> 5, kk = flat & 31;
        int gn = n0 + nn, gk = k0 + kk;
        float v = (gn < N && gk < K) ? Wm[(size_t)gn * K + gk] : 0.0f;
        Bsb[nn * LDSS + kk] = (uint16_t)bf16_1(v);
      }
    }
  } else {
    if (fullK && fullN) {
#pragma unroll
      for (int i = 0; i < 4; ++i) {
        int flat4 = i * 256 + tid;
        int nn = flat4 & 127, k4 = (flat4 >> 7) * 4;
        const float* base = Wm + (size_t)(k0 + k4) * N + (n0 + nn);
        float f0 = base[0];
        float f1 = base[(size_t)N];
        float f2 = base[(size_t)2 * N];
        float f3 = base[(size_t)3 * N];
        uint2 p = make_uint2(bf16_2(f0, f1), bf16_2(f2, f3));
        *reinterpret_cast<uint2*>(&Bsb[nn * LDSS + k4]) = p;
      }
    } else {
#pragma unroll
      for (int i = 0; i < 16; ++i) {
        int flat = i * 256 + tid;
        int kk = flat >> 7, nn = flat & 127;
        int gk = k0 + kk, gn = n0 + nn;
        float v = (gk < K && gn < N) ? Wm[(size_t)gk * N + gn] : 0.0f;
        Bsb[nn * LDSS + kk] = (uint16_t)bf16_1(v);
      }
    }
  }
}

// MODE: 0 = linear (+bias), 1 = relu (+bias), 2 = SOM distance epilogue
template <int MODE>
__global__ __launch_bounds__(256)
void gemm_wmma(const float* __restrict__ A, const float* __restrict__ Wm,
               const float* __restrict__ biasOrZ2, const float* __restrict__ P2,
               float* __restrict__ C, int M, int N, int K) {
  __shared__ uint16_t As[2][BM * LDSS];
  __shared__ uint16_t Bs[2][BN * LDSS];

  const int tid  = threadIdx.x;
  const int lane = tid & 31;
  const int wave = tid >> 5;
  const int hf   = lane >> 4;         // wave half (0/1)
  const int lr   = lane & 15;
  const int wm0  = (wave & 3) * 32;   // wave M-offset in block tile
  const int wn0  = (wave >> 2) * 64;  // wave N-offset in block tile
  const int m0   = blockIdx.y * BM;   // M is a multiple of 128 for all calls
  const int n0   = blockIdx.x * BN;

  FragC acc[2][4];
#pragma unroll
  for (int i = 0; i < 2; ++i)
#pragma unroll
    for (int j = 0; j < 4; ++j)
#pragma unroll
      for (int r = 0; r < 8; ++r) acc[i][j].f[r] = 0.0f;

  const int aOff = (wm0 + lr) * LDSS;
  const int bOff = (wn0 + lr) * LDSS;

  // prologue: stage chunk 0 into buffer 0
  stage_tiles<MODE>(A, Wm, As[0], Bs[0], tid, m0, n0, 0, N, K);
  __syncthreads();

  const int nChunks = (K + BK - 1) / BK;
  for (int kc = 0; kc < nChunks; ++kc) {
    const int cur = kc & 1;

    // stage next chunk into the other buffer first (loads issue early,
    // latency overlaps the WMMA burst below)
    if (kc + 1 < nChunks)
      stage_tiles<MODE>(A, Wm, As[cur ^ 1], Bs[cur ^ 1],
                        tid, m0, n0, (kc + 1) * BK, N, K);
    // warm GL2 two chunks ahead
    if (kc + 2 < nChunks)
      __builtin_prefetch(&A[(size_t)(m0 + (tid >> 1)) * K + (kc + 2) * BK], 0, 0);

    // ---- build fragments, contiguous k = 16*hf + s in both A and B ----
    FragAB afr[2];
#pragma unroll
    for (int i = 0; i < 2; ++i) {
      const uint16_t* ap = &As[cur][aOff + i * 16 * LDSS];
#pragma unroll
      for (int p = 0; p < 8; ++p) {
        int k = 2 * p + 16 * hf;
        afr[i].u[p] = *reinterpret_cast<const uint32_t*>(ap + k);
      }
    }
    FragAB bfr[4];
#pragma unroll
    for (int j = 0; j < 4; ++j) {
      const uint16_t* bp = &Bs[cur][bOff + j * 16 * LDSS];
#pragma unroll
      for (int p = 0; p < 8; ++p) {
        int k = 2 * p + 16 * hf;
        bfr[j].u[p] = *reinterpret_cast<const uint32_t*>(bp + k);
      }
    }
    // ---- 8 WMMAs per wave per K-chunk ----
#pragma unroll
    for (int i = 0; i < 2; ++i)
#pragma unroll
      for (int j = 0; j < 4; ++j)
        acc[i][j].v = __builtin_amdgcn_wmma_f32_16x16x32_bf16(
            false, afr[i].v, false, bfr[j].v, (short)0, acc[i][j].v,
            false, false);

    if (kc + 1 < nChunks) __syncthreads();
  }

  // ---- epilogue: bias / relu / SOM distance, predicated on N edge ----
#pragma unroll
  for (int i = 0; i < 2; ++i) {
    const int mbase = m0 + wm0 + i * 16;
#pragma unroll
    for (int j = 0; j < 4; ++j) {
      const int n = n0 + wn0 + j * 16 + lr;
      if (n < N) {
        float extra = (MODE == 2) ? P2[n] : biasOrZ2[n];
#pragma unroll
        for (int r = 0; r < 8; ++r) {
          const int m = mbase + r + 8 * hf;
          float v = acc[i][j].f[r];
          if (MODE == 2) {
            v = biasOrZ2[m] + extra - 2.0f * v;   // ||z||^2 + ||p||^2 - 2 z.p
            v = fmaxf(v, 0.0f);
          } else {
            v += extra;
            if (MODE == 1) v = fmaxf(v, 0.0f);
          }
          C[(size_t)m * N + n] = v;
        }
      }
    }
  }
}

// One wave32 per row: out[r] = sum_c X[r][c]^2
__global__ __launch_bounds__(256)
void rowsumsq(const float* __restrict__ X, float* __restrict__ out,
              int rows, int cols) {
  int warp = (int)((blockIdx.x * blockDim.x + threadIdx.x) >> 5);
  int lane = threadIdx.x & 31;
  if (warp >= rows) return;
  const float* rp = X + (size_t)warp * cols;
  float s = 0.0f;
  for (int c = lane; c < cols; c += 32) {
    float v = rp[c];
    s += v * v;
  }
#pragma unroll
  for (int off = 16; off > 0; off >>= 1) s += __shfl_xor(s, off, 32);
  if (lane == 0) out[warp] = s;
}

extern "C" void kernel_launch(void* const* d_in, const int* in_sizes, int n_in,
                              void* d_out, int out_size, void* d_ws, size_t ws_size,
                              hipStream_t stream) {
  (void)in_sizes; (void)n_in; (void)out_size; (void)ws_size;

  const float* x   = (const float*)d_in[0];
  const float* eW0 = (const float*)d_in[1];  const float* eb0 = (const float*)d_in[2];
  const float* eW1 = (const float*)d_in[3];  const float* eb1 = (const float*)d_in[4];
  const float* eW2 = (const float*)d_in[5];  const float* eb2 = (const float*)d_in[6];
  const float* eW3 = (const float*)d_in[7];  const float* eb3 = (const float*)d_in[8];
  const float* dW3 = (const float*)d_in[9];  const float* db3 = (const float*)d_in[10];
  const float* dW2 = (const float*)d_in[11]; const float* db2 = (const float*)d_in[12];
  const float* dW1 = (const float*)d_in[13]; const float* db1 = (const float*)d_in[14];
  const float* dW0 = (const float*)d_in[15]; const float* db0 = (const float*)d_in[16];
  const float* prot = (const float*)d_in[17];

  const int B = 16384;
  // workspace layout (floats)
  float* bufA = (float*)d_ws;
  float* bufB = bufA + (size_t)B * 2000;
  float* zbuf = bufB + (size_t)B * 2000;
  float* z2   = zbuf + (size_t)B * 256;
  float* p2   = z2 + B;

  float* decoded = (float*)d_out;
  float* dists   = decoded + (size_t)B * 784;

  dim3 blk(256);
  auto grid = [](int M, int N) { return dim3((unsigned)((N + 127) / 128),
                                             (unsigned)(M / 128)); };

  // encoder
  gemm_wmma<1><<<grid(B, 2000), blk, 0, stream>>>(x,    eW0, eb0, nullptr, bufA, B, 2000, 784);
  gemm_wmma<1><<<grid(B, 2000), blk, 0, stream>>>(bufA, eW1, eb1, nullptr, bufB, B, 2000, 2000);
  gemm_wmma<1><<<grid(B,  500), blk, 0, stream>>>(bufB, eW2, eb2, nullptr, bufA, B,  500, 2000);
  gemm_wmma<0><<<grid(B,  256), blk, 0, stream>>>(bufA, eW3, eb3, nullptr, zbuf, B,  256, 500);
  // decoder
  gemm_wmma<1><<<grid(B,  500), blk, 0, stream>>>(zbuf, dW3, db3, nullptr, bufA, B,  500, 256);
  gemm_wmma<1><<<grid(B, 2000), blk, 0, stream>>>(bufA, dW2, db2, nullptr, bufB, B, 2000, 500);
  gemm_wmma<1><<<grid(B, 2000), blk, 0, stream>>>(bufB, dW1, db1, nullptr, bufA, B, 2000, 2000);
  gemm_wmma<0><<<grid(B,  784), blk, 0, stream>>>(bufA, dW0, db0, nullptr, decoded, B, 784, 2000);
  // SOM: squared norms, then distance GEMM
  rowsumsq<<<dim3((B * 32 + 255) / 256), blk, 0, stream>>>(zbuf, z2, B, 256);
  rowsumsq<<<dim3((1024 * 32 + 255) / 256), blk, 0, stream>>>(prot, p2, 1024, 256);
  gemm_wmma<2><<<grid(B, 1024), blk, 0, stream>>>(zbuf, prot, z2, p2, dists, B, 1024, 256);
}
